// DiscriminatorLatent_68813966016945
// MI455X (gfx1250) — matrix operand order repaired and checked by reading, not compiled, not run on hardware
//
#include <hip/hip_runtime.h>

typedef unsigned short ushort_t;
typedef __attribute__((ext_vector_type(16))) __bf16 v16bf;
typedef __attribute__((ext_vector_type(8)))  float  v8f;

#define B_ROWS 8192
#define WIDTH  512
#define DEPTH  7
#define CIN    4096
#define EPSV   1e-5f
#define SLOPE  0.01f
#define LDSK   40   // padded LDS K-stride (bf16 elems) to spread banks

__device__ __forceinline__ ushort_t f2bf(float f) {
    unsigned int u = __float_as_uint(f);
    u += 0x7FFFu + ((u >> 16) & 1u);          // round-to-nearest-even
    return (ushort_t)(u >> 16);
}
__device__ __forceinline__ float bf2f(ushort_t h) {
    return __uint_as_float(((unsigned int)h) << 16);
}

// gfx1250 async DMA: global -> LDS, tracked by ASYNCcnt. INST_OFFSET is added
// to BOTH the LDS address and the global address, so one (lds, global) base
// pair covers a 32B chunk with offset:0 / offset:16 beats.
__device__ __forceinline__ void async_copy_32B(uint32_t lds_off, const ushort_t* gptr) {
    asm volatile("global_load_async_to_lds_b128 %0, %1, off"
                 :: "v"(lds_off), "v"(gptr) : "memory");
    asm volatile("global_load_async_to_lds_b128 %0, %1, off offset:16"
                 :: "v"(lds_off), "v"(gptr) : "memory");
}
__device__ __forceinline__ void wait_async0() {
    asm volatile("s_wait_asynccnt 0x0" ::: "memory");
}
__device__ __forceinline__ uint32_t lds_addr32(const void* p) {
    // low 32 bits of the shared-aperture flat address == wave-relative LDS offset
    return (uint32_t)(uintptr_t)p;
}

// ---------------- prep kernels ----------------
__global__ void copy_x_kernel(const float* __restrict__ x, ushort_t* __restrict__ lastA) {
    int i = blockIdx.x * blockDim.x + threadIdx.x;       // 8192*512 threads
    int b = i >> 9, c = i & 511;
    lastA[(size_t)b * CIN + c] = f2bf(x[i]);
}

__global__ void conv_f32_bf16(const float* __restrict__ src, ushort_t* __restrict__ dst, int n) {
    int i = blockIdx.x * blockDim.x + threadIdx.x;
    if (i < n) dst[i] = f2bf(src[i]);
}

__global__ void zero_kernel(float* __restrict__ p, int n) {
    int i = blockIdx.x * blockDim.x + threadIdx.x;
    if (i < n) p[i] = 0.0f;
}

// ---------------- GEMM + bias + BN partial stats ----------------
// C tile 128x128 per block; 8 waves in 4(M) x 2(N) grid; wave tile 32x64.
// Double-buffered LDS, async global->LDS DMA overlapped with WMMA compute.
__global__ __launch_bounds__(256) void gemm_bn_stats(
    const ushort_t* __restrict__ A,     // lastA [8192][4096] bf16
    const ushort_t* __restrict__ Wb,    // [512][fan_in] bf16 (K-contiguous rows)
    const float*   __restrict__ bias,   // [512]
    float*         __restrict__ H,      // [8192][512] fp32
    float*         __restrict__ sum1,   // [512]
    float*         __restrict__ sum2,   // [512]
    int fan_in)
{
    __shared__ ushort_t Al[2][128 * LDSK];
    __shared__ ushort_t Bl[2][128 * LDSK];

    const int tid  = threadIdx.x;
    const int wave = tid >> 5;
    const int lane = tid & 31;
    const int hh   = lane >> 4;      // half of wave (0/1)
    const int idx  = lane & 15;
    const int wm   = wave & 3;       // 4 waves along M
    const int wn   = wave >> 2;      // 2 waves along N
    const int rowBase = blockIdx.x * 128;
    const int colBase = blockIdx.y * 128;

    v8f acc[2][4] = {};

    // cooperative DMA issue: 2 threads per tile row, 32B (16 bf16) each
    const int ldr = tid >> 1;              // 0..127
    const int ldh = (tid & 1) * 16;        // 0 or 16
    const ushort_t* aBase = A  + (size_t)(rowBase + ldr) * CIN    + ldh;
    const ushort_t* bBase = Wb + (size_t)(colBase + ldr) * fan_in + ldh;
    const uint32_t laBase[2] = { lds_addr32(&Al[0][ldr * LDSK + ldh]),
                                 lds_addr32(&Al[1][ldr * LDSK + ldh]) };
    const uint32_t lbBase[2] = { lds_addr32(&Bl[0][ldr * LDSK + ldh]),
                                 lds_addr32(&Bl[1][ldr * LDSK + ldh]) };

    union Frag { uint4 u[2]; v16bf v; };

    const int ksteps = fan_in >> 5;

    // prologue: DMA tile 0 into buffer 0
    async_copy_32B(laBase[0], aBase);
    async_copy_32B(lbBase[0], bBase);

    for (int kt = 0; kt < ksteps; ++kt) {
        const int cur = kt & 1;
        wait_async0();          // this wave's DMAs for buffer `cur` landed
        __syncthreads();        // everyone's DMAs landed; prior reads of ~cur done

        if (kt + 1 < ksteps) {  // overlap DMA of next tile with compute
            async_copy_32B(laBase[cur ^ 1], aBase + (kt + 1) * 32);
            async_copy_32B(lbBase[cur ^ 1], bBase + (kt + 1) * 32);
        }
        if (kt + 2 < ksteps) {  // keep async engine hitting L2 (global_prefetch_b8)
            __builtin_prefetch(aBase + (kt + 2) * 32, 0, 1);
            __builtin_prefetch(bBase + (kt + 2) * 32, 0, 1);
        }

        // A fragments: ISA 16-bit A 16x32 layout -> per lane two 8-elem K runs
        Frag af[2];
        #pragma unroll
        for (int m = 0; m < 2; ++m) {
            int r = wm * 32 + m * 16 + idx;
            af[m].u[0] = *(const uint4*)&Al[cur][r * LDSK + 8 * hh];
            af[m].u[1] = *(const uint4*)&Al[cur][r * LDSK + 16 + 8 * hh];
        }
        // B fragments: per lane two 8-elem K runs, same layout as A
        Frag bfr[4];
        #pragma unroll
        for (int n = 0; n < 4; ++n) {
            int c = wn * 64 + n * 16 + idx;
            bfr[n].u[0] = *(const uint4*)&Bl[cur][c * LDSK + 8 * hh];
            bfr[n].u[1] = *(const uint4*)&Bl[cur][c * LDSK + 16 + 8 * hh];
        }

        #pragma unroll
        for (int m = 0; m < 2; ++m)
            #pragma unroll
            for (int n = 0; n < 4; ++n)
                acc[m][n] = __builtin_amdgcn_wmma_f32_16x16x32_bf16(
                    false, af[m].v, false, bfr[n].v, (short)0, acc[m][n], false, false);
    }

    // epilogue: bias add, store H, per-column partial sums for batchnorm
    #pragma unroll
    for (int n = 0; n < 4; ++n) {
        int col = colBase + wn * 64 + n * 16 + idx;
        float bcol = bias[col];
        float s1 = 0.0f, s2 = 0.0f;
        #pragma unroll
        for (int m = 0; m < 2; ++m) {
            int row0 = rowBase + wm * 32 + m * 16 + 8 * hh;
            #pragma unroll
            for (int r = 0; r < 8; ++r) {
                float v = acc[m][n][r] + bcol;
                H[(size_t)(row0 + r) * WIDTH + col] = v;
                s1 += v;
                s2 += v * v;
            }
        }
        s1 += __shfl_xor(s1, 16, 32);
        s2 += __shfl_xor(s2, 16, 32);
        if (hh == 0) {
            atomicAdd(&sum1[col], s1);
            atomicAdd(&sum2[col], s2);
        }
    }
}

// ---------------- BN finalize + elementwise ----------------
__global__ void finalize_stats(const float* __restrict__ sum1, const float* __restrict__ sum2,
                               const float* __restrict__ gamma, const float* __restrict__ beta,
                               float* __restrict__ scale, float* __restrict__ shift)
{
    int c = threadIdx.x;   // 512
    float mean = sum1[c] * (1.0f / B_ROWS);
    float var  = sum2[c] * (1.0f / B_ROWS) - mean * mean;   // biased variance
    float sc   = gamma[c] * rsqrtf(var + EPSV);
    scale[c] = sc;
    shift[c] = beta[c] - mean * sc;
}

__global__ void bn_act(const float* __restrict__ H, const float* __restrict__ scale,
                       const float* __restrict__ shift, const float* __restrict__ noise,
                       ushort_t* __restrict__ lastA, int off)
{
    int i = blockIdx.x * blockDim.x + threadIdx.x;   // 8192*512
    int b = i >> 9, c = i & 511;
    float y = H[i] * scale[c] + shift[c];
    y = (y >= 0.0f) ? y : SLOPE * y;
    y *= noise[i];
    lastA[(size_t)b * CIN + off + c] = f2bf(y);
}

// ---------------- final projection + sigmoid ----------------
__global__ __launch_bounds__(256) void final_proj(const ushort_t* __restrict__ lastA,
                                                  const float* __restrict__ Wc,
                                                  const float* __restrict__ bc,
                                                  float* __restrict__ out)
{
    int wave = threadIdx.x >> 5;
    int lane = threadIdx.x & 31;
    int row  = blockIdx.x * 8 + wave;
    const ushort_t* a = lastA + (size_t)row * CIN;
    float s = 0.0f;
    #pragma unroll 4
    for (int j = 0; j < CIN / 32; ++j) {
        int c = j * 32 + lane;
        s += bf2f(a[c]) * Wc[c];
    }
    #pragma unroll
    for (int o = 16; o > 0; o >>= 1) s += __shfl_xor(s, o, 32);
    if (lane == 0) out[row] = 1.0f / (1.0f + __expf(-(s + bc[0])));
}

// ---------------- host ----------------
extern "C" void kernel_launch(void* const* d_in, const int* in_sizes, int n_in,
                              void* d_out, int out_size, void* d_ws, size_t ws_size,
                              hipStream_t stream)
{
    (void)in_sizes; (void)n_in; (void)out_size; (void)ws_size;
    const float* x     = (const float*)d_in[0];
    const float* W[7];
    for (int i = 0; i < 7; ++i) W[i] = (const float*)d_in[1 + i];
    const float* b     = (const float*)d_in[8];
    const float* gamma = (const float*)d_in[9];
    const float* beta  = (const float*)d_in[10];
    const float* Wc    = (const float*)d_in[11];
    const float* bc    = (const float*)d_in[12];
    const float* noise = (const float*)d_in[13];
    float* out = (float*)d_out;

    char* ws = (char*)d_ws;
    ushort_t* lastA = (ushort_t*)ws;                          // 8192*4096 bf16 = 67,108,864 B
    float*    H     = (float*)(ws + 67108864);                // 8192*512 f32  = 16,777,216 B
    ushort_t* Wb    = (ushort_t*)(ws + 83886080);             // 7,340,032 bf16 = 14,680,064 B
    float*    sum1  = (float*)(ws + 98566144);                // 512
    float*    sum2  = sum1 + 512;                             // 512
    float*    scale = sum1 + 1024;                            // 512
    float*    shift = sum1 + 1536;                            // 512

    // bf16 conversions
    copy_x_kernel<<<(B_ROWS * 512) / 256, 256, 0, stream>>>(x, lastA);
    ushort_t* WbPtr[7];
    size_t woff = 0;
    for (int i = 0; i < 7; ++i) {
        int fan = 512 * (i + 1);
        int n = WIDTH * fan;
        WbPtr[i] = Wb + woff;
        conv_f32_bf16<<<(n + 255) / 256, 256, 0, stream>>>(W[i], WbPtr[i], n);
        woff += (size_t)n;
    }

    // layers
    for (int i = 0; i < 7; ++i) {
        int fan = 512 * (i + 1);
        zero_kernel<<<1, 1024, 0, stream>>>(sum1, 1024);   // clears sum1+sum2
        gemm_bn_stats<<<dim3(B_ROWS / 128, WIDTH / 128), 256, 0, stream>>>(
            lastA, WbPtr[i], b + i * 512, H, sum1, sum2, fan);
        finalize_stats<<<1, 512, 0, stream>>>(sum1, sum2, gamma + i * 512, beta + i * 512,
                                              scale, shift);
        bn_act<<<(B_ROWS * 512) / 256, 256, 0, stream>>>(
            H, scale, shift, noise + (size_t)i * B_ROWS * 512, lastA, 512 + i * 512);
    }

    final_proj<<<B_ROWS / 8, 256, 0, stream>>>(lastA, Wc, bc, out);
}
